// MultiHeadAttention_47227460387052
// MI455X (gfx1250) — matrix operand order repaired and checked by reading, not compiled, not run on hardware
//
#include <hip/hip_runtime.h>
#include <hip/hip_bf16.h>
#include <stdint.h>

typedef __bf16 bf16;
typedef __bf16 v16bf __attribute__((ext_vector_type(16)));
typedef __bf16 v8bf  __attribute__((ext_vector_type(8)));
typedef float  v8f   __attribute__((ext_vector_type(8)));
typedef float  v4f   __attribute__((ext_vector_type(4)));
typedef unsigned int v4u  __attribute__((ext_vector_type(4)));
typedef int          v8i_ __attribute__((ext_vector_type(8)));
typedef int          v4i_ __attribute__((ext_vector_type(4)));

#define B_  2
#define S_  2048
#define D_  1024
#define H_  16
#define DK_ 64

union BF16x16 { v16bf v; v8bf h[2]; };

// Async global->LDS 16B copy (GLOBAL_LOAD_ASYNC_TO_LDS_B128, ASYNCcnt).
__device__ __forceinline__ void async_copy16(const bf16* lds_dst, const bf16* gsrc) {
  unsigned la = (unsigned)(uintptr_t)lds_dst;   // low 32 bits = LDS byte addr
  asm volatile("global_load_async_to_lds_b128 %0, %1, off"
               :: "v"(la), "v"(gsrc) : "memory");
}
__device__ __forceinline__ void wait_async0() {
  asm volatile("s_wait_asynccnt 0x0" ::: "memory");
}

// TDM: DMA a [rows x 64] bf16 tile (rows contiguous, 128B each) into LDS with
// 16B padding after every row (pad_interval=32 DW -> code 4, pad_amount=4 DW
// -> code 3) => padded LDS row stride 144B = 72 halves.
__device__ __forceinline__ void tdm_load_tile_64x128(const bf16* lds_dst,
                                                     const bf16* gsrc) {
  unsigned lds = (unsigned)(uintptr_t)lds_dst;
  unsigned long long ga = (unsigned long long)(uintptr_t)gsrc;
  v4u g0;
  g0[0] = 1u;                                   // count=1 (valid descriptor)
  g0[1] = lds;                                  // lds_addr
  g0[2] = (unsigned)ga;                         // global_addr[31:0]
  g0[3] = (unsigned)((ga >> 32) & 0x01FFFFFFu)  // global_addr[56:32]
        | (2u << 30);                           // type=2 ("image")
  v8i_ g1;
  g1[0] = (int)((1u << 16)    // data_size = 1 -> 2 bytes
              | (1u << 20)    // pad_enable
              | (4u << 22)    // pad_interval: 32 DWORDs
              | (3u << 25));  // pad_amount: 4 DWORDs (16B)
  g1[1] = (int)(64u << 16);   // tensor_dim0[15:0] = 64
  g1[2] = (int)((unsigned)S_ << 16); // tensor_dim1[15:0] = 2048
  g1[3] = (int)(64u << 16);   // tile_dim0 = 64
  g1[4] = 128;                // tile_dim1 = 128 rows
  g1[5] = 64;                 // tensor_dim0_stride = 64 elements
  g1[6] = 0;
  g1[7] = 0;
  v4i_ gz4 = {0, 0, 0, 0};
  v8i_ gz8 = {0, 0, 0, 0, 0, 0, 0, 0};
  // clang-23 / amdgpu-toolchain 6-arg form:
  // (g0 v4u, g1 v8i, g2 v4i, g3 v4i, v8i, cpol)
  __builtin_amdgcn_tensor_load_to_lds(g0, g1, gz4, gz4, gz8, 0);
}
__device__ __forceinline__ void wait_tensor0() {
  __builtin_amdgcn_s_wait_tensorcnt((short)0);
}

// A-fragment (16x32 bf16, MxK): lane L -> row M = L%16.
__device__ __forceinline__ v16bf ldsA_frag(const bf16* As, int stride, int m0, int k0, int lane) {
  const int row = lane & 15, hi = lane >> 4;
  const bf16* base = As + (m0 + row) * stride + k0 + hi * 8;
  BF16x16 u;
  u.h[0] = *(const v8bf*)(base);
  u.h[1] = *(const v8bf*)(base + 16);
  return u.v;
}

// B-fragment (32x16 bf16, KxN) from LDS stored as Bs[n][k].
__device__ __forceinline__ v16bf ldsB_frag(const bf16* Bs, int stride, int n0, int k0, int lane) {
  const int col = lane & 15, hi = lane >> 4;
  const bf16* base = Bs + (n0 + col) * stride + k0 + hi * 8;
  BF16x16 u;
  u.h[0] = *(const v8bf*)(base);
  u.h[1] = *(const v8bf*)(base + 16);
  return u.v;
}

#define WMMA_BF16(a, b, c) \
  __builtin_amdgcn_wmma_f32_16x16x32_bf16(false, (a), false, (b), (short)0, (c), false, false)

// ---------------------------------------------------------------------------
// Projection: Y[m,n] = sum_k X[m,k]*W[n,k] + bias[n]; bf16 out in [B,H,S,DK].
// Double-buffered LDS: stage tile t+1 while WMMA consumes tile t.
// ---------------------------------------------------------------------------
__global__ __launch_bounds__(256) void proj_kernel(
    const float* __restrict__ X, const float* __restrict__ W,
    const float* __restrict__ bias, bf16* __restrict__ outH) {
  constexpr int SA = 40;
  __shared__ bf16 As0[128 * SA], Bs0[128 * SA];
  __shared__ bf16 As1[128 * SA], Bs1[128 * SA];
  const int t = threadIdx.x, lane = t & 31, wid = t >> 5;
  const int mt = blockIdx.y * 128, nt = blockIdx.x * 128;
  const int wm = (wid >> 2) * 64, wn = (wid & 3) * 32;

  v8f acc[4][2];
  const v8f vzero = {0.f,0.f,0.f,0.f,0.f,0.f,0.f,0.f};
#pragma unroll
  for (int i = 0; i < 4; ++i)
#pragma unroll
    for (int j = 0; j < 2; ++j) acc[i][j] = vzero;

  auto load_tile = [&](int kk, v4f* xa, v4f* wa) {
#pragma unroll
    for (int i = 0; i < 4; ++i) {
      int e = t + i * 256;
      int row = e >> 3, c4 = (e & 7) << 2;
      xa[i] = *(const v4f*)(X + (size_t)(mt + row) * D_ + kk + c4);
      wa[i] = *(const v4f*)(W + (size_t)(nt + row) * D_ + kk + c4);
    }
    if (kk + 32 < D_) {
      __builtin_prefetch(X + (size_t)(mt + (t >> 1)) * D_ + kk + 32);
      __builtin_prefetch(W + (size_t)(nt + (t >> 1)) * D_ + kk + 32);
    }
  };
  auto store_tile = [&](const v4f* xa, const v4f* wa, bf16* A, bf16* Bb) {
#pragma unroll
    for (int i = 0; i < 4; ++i) {
      int e = t + i * 256;
      int row = e >> 3, c4 = (e & 7) << 2;
      bf16* da = A + row * SA + c4;
      da[0] = (bf16)xa[i][0]; da[1] = (bf16)xa[i][1];
      da[2] = (bf16)xa[i][2]; da[3] = (bf16)xa[i][3];
      bf16* db = Bb + row * SA + c4;
      db[0] = (bf16)wa[i][0]; db[1] = (bf16)wa[i][1];
      db[2] = (bf16)wa[i][2]; db[3] = (bf16)wa[i][3];
    }
  };
  auto compute = [&](const bf16* A, const bf16* Bb) {
    v16bf af[4], bfr[2];
#pragma unroll
    for (int ms = 0; ms < 4; ++ms) af[ms] = ldsA_frag(A, SA, wm + ms * 16, 0, lane);
#pragma unroll
    for (int ns = 0; ns < 2; ++ns) bfr[ns] = ldsB_frag(Bb, SA, wn + ns * 16, 0, lane);
#pragma unroll
    for (int ms = 0; ms < 4; ++ms)
#pragma unroll
      for (int ns = 0; ns < 2; ++ns) acc[ms][ns] = WMMA_BF16(af[ms], bfr[ns], acc[ms][ns]);
  };

  { v4f xa[4], wa[4]; load_tile(0, xa, wa); store_tile(xa, wa, As0, Bs0); }
  __syncthreads();
  for (int kk = 0; kk < D_; kk += 64) {
    {
      v4f xa[4], wa[4];
      load_tile(kk + 32, xa, wa);        // in flight under the WMMAs
      compute(As0, Bs0);
      store_tile(xa, wa, As1, Bs1);
      __syncthreads();
    }
    {
      v4f xa[4], wa[4];
      const bool more = (kk + 64 < D_);
      if (more) load_tile(kk + 64, xa, wa);
      compute(As1, Bs1);
      if (more) store_tile(xa, wa, As0, Bs0);
      __syncthreads();
    }
  }

  const int col = lane & 15, rhi = (lane >> 4) * 8;
#pragma unroll
  for (int ms = 0; ms < 4; ++ms)
#pragma unroll
    for (int ns = 0; ns < 2; ++ns) {
      int n = nt + wn + ns * 16 + col;
      float bv = bias[n];
      int hh = n >> 6, dk = n & 63;
#pragma unroll
      for (int r = 0; r < 8; ++r) {
        int m = mt + wm + ms * 16 + r + rhi;
        int b = m / S_, s = m % S_;
        float val = acc[ms][ns][r] + bv;
        outH[(((size_t)b * H_ + hh) * S_ + s) * DK_ + dk] = (bf16)val;
      }
    }
}

// ---------------------------------------------------------------------------
// Scores: S[q,k] = (qh . kh) / 8, masked -> fp32 scratch.
// Tiles staged by the Tensor Data Mover (hardware-padded LDS rows).
// ---------------------------------------------------------------------------
__global__ __launch_bounds__(256) void scores_kernel(
    const bf16* __restrict__ qh, const bf16* __restrict__ kh,
    const unsigned char* __restrict__ mask, float* __restrict__ attn) {
  constexpr int SA = 72;                 // 64 data + 8 pad halves (TDM pad)
  __shared__ bf16 As[128 * SA];
  __shared__ bf16 Bs[128 * SA];
  const int t = threadIdx.x, lane = t & 31, wid = t >> 5;
  const int nt = blockIdx.x * 128, mt = blockIdx.y * 128;
  const int bh = blockIdx.z, b = bh >> 4;
  const bf16* qb = qh + ((size_t)bh * S_ + mt) * DK_;
  const bf16* kb = kh + ((size_t)bh * S_ + nt) * DK_;

  if (wid == 0) {                        // one wave drives the TDM
    tdm_load_tile_64x128(As, qb);
    tdm_load_tile_64x128(Bs, kb);
    wait_tensor0();
  }
  __syncthreads();

  const int wm = (wid >> 2) * 64, wn = (wid & 3) * 32;
  v8f acc[4][2];
  const v8f vzero = {0.f,0.f,0.f,0.f,0.f,0.f,0.f,0.f};
#pragma unroll
  for (int i = 0; i < 4; ++i)
#pragma unroll
    for (int j = 0; j < 2; ++j) acc[i][j] = vzero;

#pragma unroll
  for (int k0 = 0; k0 < DK_; k0 += 32) {
    v16bf af[4], bfr[2];
#pragma unroll
    for (int ms = 0; ms < 4; ++ms) af[ms] = ldsA_frag(As, SA, wm + ms * 16, k0, lane);
#pragma unroll
    for (int ns = 0; ns < 2; ++ns) bfr[ns] = ldsB_frag(Bs, SA, wn + ns * 16, k0, lane);
#pragma unroll
    for (int ms = 0; ms < 4; ++ms)
#pragma unroll
      for (int ns = 0; ns < 2; ++ns) acc[ms][ns] = WMMA_BF16(af[ms], bfr[ns], acc[ms][ns]);
  }

  const int col = lane & 15, rhi = (lane >> 4) * 8;
#pragma unroll
  for (int ms = 0; ms < 4; ++ms)
#pragma unroll
    for (int ns = 0; ns < 2; ++ns) {
      int kc = nt + wn + ns * 16 + col;
#pragma unroll
      for (int r = 0; r < 8; ++r) {
        int q = mt + wm + ms * 16 + r + rhi;
        float v = acc[ms][ns][r] * 0.125f;   // 1/sqrt(64)
        unsigned char mv = mask[((size_t)b * S_ + q) * S_ + kc];
        attn[((size_t)bh * S_ + q) * S_ + kc] = mv ? -1e9f : v;
      }
    }
}

// ---------------------------------------------------------------------------
// Row softmax * S_k, in place. One block per (b,h,q) row.
// ---------------------------------------------------------------------------
__global__ __launch_bounds__(256) void softmax_kernel(float* __restrict__ attn) {
  __shared__ float red[256];
  const int t = threadIdx.x;
  float* row = attn + (size_t)blockIdx.x * S_;
  float v[8];
  float mx = -3.4e38f;
#pragma unroll
  for (int j = 0; j < 8; ++j) { v[j] = row[t + j * 256]; mx = fmaxf(mx, v[j]); }
  red[t] = mx; __syncthreads();
  for (int off = 128; off > 0; off >>= 1) {
    if (t < off) red[t] = fmaxf(red[t], red[t + off]);
    __syncthreads();
  }
  mx = red[0]; __syncthreads();
  float sum = 0.f;
#pragma unroll
  for (int j = 0; j < 8; ++j) { v[j] = __expf(v[j] - mx); sum += v[j]; }
  red[t] = sum; __syncthreads();
  for (int off = 128; off > 0; off >>= 1) {
    if (t < off) red[t] += red[t + off];
    __syncthreads();
  }
  const float scale = (float)S_ / red[0];
#pragma unroll
  for (int j = 0; j < 8; ++j) row[t + j * 256] = v[j] * scale;
}

// ---------------------------------------------------------------------------
// Context: ctx[q,dk] = sum_k attn[q,k] * v[k,dk]; double-buffered LDS.
// ---------------------------------------------------------------------------
__global__ __launch_bounds__(256) void ctx_kernel(
    const float* __restrict__ attn, const bf16* __restrict__ vh,
    bf16* __restrict__ ctx) {
  constexpr int SA = 40;
  __shared__ bf16 As0[128 * SA], As1[128 * SA];  // attn tiles [q][k]
  __shared__ bf16 Bs0[64 * SA],  Bs1[64 * SA];   // v^T tiles  [dk][k]
  const int t = threadIdx.x, lane = t & 31, wid = t >> 5;
  const int mt = blockIdx.x * 128;
  const int bh = blockIdx.y, b = bh >> 4, h = bh & 15;
  const float* ab = attn + ((size_t)bh * S_ + mt) * S_;
  const bf16* vb = vh + (size_t)bh * S_ * DK_;
  const int wm = (wid >> 2) * 64, wn = (wid & 3) * 16;

  v8f acc[4];
  const v8f vzero = {0.f,0.f,0.f,0.f,0.f,0.f,0.f,0.f};
#pragma unroll
  for (int i = 0; i < 4; ++i) acc[i] = vzero;

  auto load_tile = [&](int kk, v4f* xa, bf16* vv) {
#pragma unroll
    for (int i = 0; i < 4; ++i) {
      int e = t + i * 256;
      int row = e >> 3, c4 = (e & 7) << 2;
      xa[i] = *(const v4f*)(ab + (size_t)row * S_ + kk + c4);
    }
#pragma unroll
    for (int i = 0; i < 8; ++i) {
      int e = t + i * 256;
      int kl = e >> 6, dk = e & 63;
      vv[i] = vb[(size_t)(kk + kl) * DK_ + dk];
    }
  };
  auto store_tile = [&](const v4f* xa, const bf16* vv, bf16* A, bf16* Bb) {
#pragma unroll
    for (int i = 0; i < 4; ++i) {
      int e = t + i * 256;
      int row = e >> 3, c4 = (e & 7) << 2;
      bf16* d = A + row * SA + c4;
      d[0] = (bf16)xa[i][0]; d[1] = (bf16)xa[i][1];
      d[2] = (bf16)xa[i][2]; d[3] = (bf16)xa[i][3];
    }
#pragma unroll
    for (int i = 0; i < 8; ++i) {
      int e = t + i * 256;
      int kl = e >> 6, dk = e & 63;
      Bb[dk * SA + kl] = vv[i];          // transpose into [dk][k]
    }
  };
  auto compute = [&](const bf16* A, const bf16* Bb) {
    v16bf bfr = ldsB_frag(Bb, SA, wn, 0, lane);
#pragma unroll
    for (int ms = 0; ms < 4; ++ms) {
      v16bf af = ldsA_frag(A, SA, wm + ms * 16, 0, lane);
      acc[ms] = WMMA_BF16(af, bfr, acc[ms]);
    }
  };

  { v4f xa[4]; bf16 vv[8]; load_tile(0, xa, vv); store_tile(xa, vv, As0, Bs0); }
  __syncthreads();
  for (int kk = 0; kk < S_; kk += 64) {
    {
      v4f xa[4]; bf16 vv[8];
      load_tile(kk + 32, xa, vv);
      compute(As0, Bs0);
      store_tile(xa, vv, As1, Bs1);
      __syncthreads();
    }
    {
      v4f xa[4]; bf16 vv[8];
      const bool more = (kk + 64 < S_);
      if (more) load_tile(kk + 64, xa, vv);
      compute(As1, Bs1);
      if (more) store_tile(xa, vv, As0, Bs0);
      __syncthreads();
    }
  }

  const int col = lane & 15, rhi = (lane >> 4) * 8;
  const int dk = wn + col;
#pragma unroll
  for (int ms = 0; ms < 4; ++ms)
#pragma unroll
    for (int r = 0; r < 8; ++r) {
      int s = mt + wm + ms * 16 + r + rhi;
      ctx[((size_t)b * S_ + s) * D_ + h * DK_ + dk] = (bf16)acc[ms][r];
    }
}

// ---------------------------------------------------------------------------
// Output projection: out[m,n] = sum_k ctx[m,k]*Wo[n,k] + bo[n]  (fp32 out)
// A tile: async DMA to LDS (double-buffered); B tile: fp32->bf16 batched.
// ---------------------------------------------------------------------------
__global__ __launch_bounds__(256) void out_kernel(
    const bf16* __restrict__ ctx, const float* __restrict__ W,
    const float* __restrict__ bias, float* __restrict__ out) {
  constexpr int SA = 40;
  __shared__ bf16 As0[128 * SA], Bs0[128 * SA];
  __shared__ bf16 As1[128 * SA], Bs1[128 * SA];
  const int t = threadIdx.x, lane = t & 31, wid = t >> 5;
  const int mt = blockIdx.y * 128, nt = blockIdx.x * 128;
  const int wm = (wid >> 2) * 64, wn = (wid & 3) * 32;

  v8f acc[4][2];
  const v8f vzero = {0.f,0.f,0.f,0.f,0.f,0.f,0.f,0.f};
#pragma unroll
  for (int i = 0; i < 4; ++i)
#pragma unroll
    for (int j = 0; j < 2; ++j) acc[i][j] = vzero;

  auto stage_async_A = [&](int kk, bf16* A) {
#pragma unroll
    for (int i = 0; i < 2; ++i) {
      int e = t + i * 256;
      int row = e >> 2, c8 = (e & 3) * 8;
      async_copy16(A + row * SA + c8, ctx + (size_t)(mt + row) * D_ + kk + c8);
    }
  };
  auto load_B = [&](int kk, v4f* wa) {
#pragma unroll
    for (int i = 0; i < 4; ++i) {
      int e = t + i * 256;
      int row = e >> 3, c4 = (e & 7) << 2;
      wa[i] = *(const v4f*)(W + (size_t)(nt + row) * D_ + kk + c4);
    }
  };
  auto store_B = [&](const v4f* wa, bf16* Bb) {
#pragma unroll
    for (int i = 0; i < 4; ++i) {
      int e = t + i * 256;
      int row = e >> 3, c4 = (e & 7) << 2;
      bf16* db = Bb + row * SA + c4;
      db[0] = (bf16)wa[i][0]; db[1] = (bf16)wa[i][1];
      db[2] = (bf16)wa[i][2]; db[3] = (bf16)wa[i][3];
    }
  };
  auto compute = [&](const bf16* A, const bf16* Bb) {
    v16bf af[4], bfr[2];
#pragma unroll
    for (int ms = 0; ms < 4; ++ms) af[ms] = ldsA_frag(A, SA, wm + ms * 16, 0, lane);
#pragma unroll
    for (int ns = 0; ns < 2; ++ns) bfr[ns] = ldsB_frag(Bb, SA, wn + ns * 16, 0, lane);
#pragma unroll
    for (int ms = 0; ms < 4; ++ms)
#pragma unroll
      for (int ns = 0; ns < 2; ++ns) acc[ms][ns] = WMMA_BF16(af[ms], bfr[ns], acc[ms][ns]);
  };

  stage_async_A(0, As0);
  { v4f wa[4]; load_B(0, wa); store_B(wa, Bs0); }
  wait_async0();
  __syncthreads();
  for (int kk = 0; kk < D_; kk += 64) {
    {
      v4f wa[4];
      stage_async_A(kk + 32, As1);       // async DMA under the WMMAs
      load_B(kk + 32, wa);
      compute(As0, Bs0);
      store_B(wa, Bs1);
      wait_async0();
      __syncthreads();
    }
    {
      v4f wa[4];
      const bool more = (kk + 64 < D_);
      if (more) { stage_async_A(kk + 64, As0); load_B(kk + 64, wa); }
      compute(As1, Bs1);
      if (more) store_B(wa, Bs0);
      wait_async0();
      __syncthreads();
    }
  }

  const int col = lane & 15, rhi = (lane >> 4) * 8;
#pragma unroll
  for (int ms = 0; ms < 4; ++ms)
#pragma unroll
    for (int ns = 0; ns < 2; ++ns) {
      int n = nt + wn + ns * 16 + col;
      float bv = bias[n];
#pragma unroll
      for (int r = 0; r < 8; ++r) {
        int m = mt + wm + ms * 16 + r + rhi;
        out[(size_t)m * D_ + n] = acc[ms][ns][r] + bv;
      }
    }
}

// ---------------------------------------------------------------------------
extern "C" void kernel_launch(void* const* d_in, const int* in_sizes, int n_in,
                              void* d_out, int out_size, void* d_ws, size_t ws_size,
                              hipStream_t stream) {
  (void)in_sizes; (void)n_in; (void)out_size; (void)ws_size;
  const float* Q  = (const float*)d_in[0];
  const float* K  = (const float*)d_in[1];
  const float* V  = (const float*)d_in[2];
  const unsigned char* mask = (const unsigned char*)d_in[3];  // bool, 1 byte
  const float* Wq = (const float*)d_in[4];
  const float* bq = (const float*)d_in[5];
  const float* Wk = (const float*)d_in[6];
  const float* bk = (const float*)d_in[7];
  const float* Wv = (const float*)d_in[8];
  const float* bv = (const float*)d_in[9];
  const float* Wo = (const float*)d_in[10];
  const float* bo = (const float*)d_in[11];

  float* out  = (float*)d_out;
  float* attn = out + (size_t)B_ * S_ * D_;      // attn_weights region of d_out

  const size_t NH = (size_t)B_ * H_ * S_ * DK_;  // 4,194,304 bf16 elems
  bf16* qh   = (bf16*)d_ws;                      // [B,H,S,DK] bf16
  bf16* kh   = qh + NH;
  bf16* vh   = kh + NH;
  bf16* ctxb = vh + NH;                          // [B,S,D] bf16

  dim3 blk(256);
  proj_kernel<<<dim3(8, 32), blk, 0, stream>>>(Q, Wq, bq, qh);
  proj_kernel<<<dim3(8, 32), blk, 0, stream>>>(K, Wk, bk, kh);
  proj_kernel<<<dim3(8, 32), blk, 0, stream>>>(V, Wv, bv, vh);
  scores_kernel<<<dim3(16, 16, 32), blk, 0, stream>>>(qh, kh, mask, attn);
  softmax_kernel<<<dim3(B_ * H_ * S_), blk, 0, stream>>>(attn);
  ctx_kernel<<<dim3(16, 32), blk, 0, stream>>>(attn, vh, ctxb);
  out_kernel<<<dim3(8, 32), blk, 0, stream>>>(ctxb, Wo, bo, out);
}
